// SelfAttention_29850022707260
// MI455X (gfx1250) — compile-verified, matrix operand order checked by reading, not run on hardware
//
#include <hip/hip_runtime.h>
#include <hip/hip_bf16.h>

#define BB  4
#define CCH 256
#define NN  4096
#define DQ  32

typedef __bf16 bf16;
typedef __attribute__((ext_vector_type(16))) __bf16 v16bf;
typedef __attribute__((ext_vector_type(8)))  __bf16 v8bf;
typedef __attribute__((ext_vector_type(8)))  float  v8f;
typedef int v4i_vs __attribute__((vector_size(16)));

union BF16x16 { v16bf v; v8bf h[2]; __bf16 e[16]; };

__device__ __forceinline__ v8f wmma_bf16(v16bf a, v16bf b, v8f c) {
  return __builtin_amdgcn_wmma_f32_16x16x32_bf16(false, a, false, b, (short)0, c,
                                                 false, false);
}

__device__ __forceinline__ float hmax16(float v) {
  v = fmaxf(v, __shfl_xor(v, 1));
  v = fmaxf(v, __shfl_xor(v, 2));
  v = fmaxf(v, __shfl_xor(v, 4));
  v = fmaxf(v, __shfl_xor(v, 8));
  return v;
}
__device__ __forceinline__ float hsum16(float v) {
  v += __shfl_xor(v, 1);
  v += __shfl_xor(v, 2);
  v += __shfl_xor(v, 4);
  v += __shfl_xor(v, 8);
  return v;
}

// Async global->LDS staging (ASYNCcnt-tracked) with plain-LDS fallback.
// Builtin signature (from hipcc diagnostic): (v4i AS1*, v4i AS3*, imm, imm).
__device__ __forceinline__ void stage16(const bf16* g, void* l) {
#if defined(__gfx1250__) && __has_builtin(__builtin_amdgcn_global_load_async_to_lds_b128)
  __builtin_amdgcn_global_load_async_to_lds_b128(
      (__attribute__((address_space(1))) v4i_vs*)(bf16*)g,
      (__attribute__((address_space(3))) v4i_vs*)l, 0, 0);
#else
  *(v8bf*)l = *(const v8bf*)g;
#endif
}
__device__ __forceinline__ void wait_stage() {
#if defined(__gfx1250__) && __has_builtin(__builtin_amdgcn_global_load_async_to_lds_b128)
#if __has_builtin(__builtin_amdgcn_s_wait_asynccnt)
  __builtin_amdgcn_s_wait_asynccnt(0);
#else
  asm volatile("s_wait_asynccnt 0" ::: "memory");
#endif
#else
  asm volatile("s_wait_dscnt 0" ::: "memory");
#endif
}

// ---------------------------------------------------------------------------
// Projection GEMM: out = W @ x per batch as 16(n) x 16(o) WMMA tiles, K=C.
// TR=0: out[b][n][O] (Q/K);  TR=1: out[b][O][n] (V).
// 16-bit A/B fragment layouts: lane<16 holds K in {0..7,16..23}, lane>=16
// holds K in {8..15,24..31}; packed pair v holds K = 2(v&3)+8*hi+16*(v>>2).
// ---------------------------------------------------------------------------
template <int TR>
__global__ __launch_bounds__(256) void proj_kernel(
    const float* __restrict__ x,     // [B][C][N]
    const float* __restrict__ W,     // [O][C]
    const float* __restrict__ bias,  // [O]
    bf16* __restrict__ out,
    int O, int total_tiles)
{
  const int wid = blockIdx.x * (blockDim.x >> 5) + (threadIdx.x >> 5);
  if (wid >= total_tiles) return;                 // wave-uniform
  const int lane = threadIdx.x & 31;
  const int lq = lane & 15, hi = lane >> 4;

  const int otiles = O >> 4;
  const int ot = wid % otiles;
  const int nt = (wid / otiles) & 255;            // N/16 == 256
  const int b  = wid / (otiles * 256);
  const int n0 = nt * 16, o0 = ot * 16;

  const float* xb   = x + (size_t)b * CCH * NN + (n0 + lq);
  const float* wrow = W + (size_t)(o0 + lq) * CCH;

  v8f acc = {};
  for (int c0 = 0; c0 < CCH; c0 += 32) {
    BF16x16 af, bf;
    // B fragment: weight row is contiguous along K -> four b128 loads
    const float4 wa = *(const float4*)(wrow + c0 + 8 * hi);
    const float4 wb = *(const float4*)(wrow + c0 + 8 * hi + 4);
    const float4 wc = *(const float4*)(wrow + c0 + 16 + 8 * hi);
    const float4 wd = *(const float4*)(wrow + c0 + 16 + 8 * hi + 4);
    bf.e[0] = (__bf16)wa.x;  bf.e[1] = (__bf16)wa.y;
    bf.e[2] = (__bf16)wa.z;  bf.e[3] = (__bf16)wa.w;
    bf.e[4] = (__bf16)wb.x;  bf.e[5] = (__bf16)wb.y;
    bf.e[6] = (__bf16)wb.z;  bf.e[7] = (__bf16)wb.w;
    bf.e[8]  = (__bf16)wc.x; bf.e[9]  = (__bf16)wc.y;
    bf.e[10] = (__bf16)wc.z; bf.e[11] = (__bf16)wc.w;
    bf.e[12] = (__bf16)wd.x; bf.e[13] = (__bf16)wd.y;
    bf.e[14] = (__bf16)wd.z; bf.e[15] = (__bf16)wd.w;
    // A fragment: x rows strided by N (coalesced across the 16-lane group)
    #pragma unroll
    for (int v = 0; v < 8; ++v) {
      const int kk = c0 + 2 * (v & 3) + 8 * hi + 16 * (v >> 2);
      af.e[2*v]   = (__bf16)xb[(size_t)kk * NN];
      af.e[2*v+1] = (__bf16)xb[(size_t)(kk + 1) * NN];
    }
    acc = wmma_bf16(af.v, bf.v, acc);
  }

  const float bval = bias[o0 + lq];
  // C-layout: lane holds rows M = r + 8*hi at column lq.
  if (TR == 0) {
    bf16* o = out + ((size_t)b * NN + n0 + 8 * hi) * O + (o0 + lq);
    #pragma unroll
    for (int r = 0; r < 8; ++r) o[(size_t)r * O] = (__bf16)(acc[r] + bval);
  } else {
    bf16* o = out + ((size_t)b * O + o0 + lq) * NN + (n0 + 8 * hi);
    #pragma unroll
    for (int r = 0; r < 8; ++r) o[r] = (__bf16)(acc[r] + bval);
  }
}

// ---------------------------------------------------------------------------
// Two-pass flash attention. Each wave owns one 16-query tile; 4 waves/block
// share a batch and cooperatively async-stage the V tile in LDS.
// Pass 1: row maxima via QK^T WMMAs only (K is L2-resident: 256 KB/batch).
// Pass 2: P = exp2(S*sc - m*sc), accumulate l and O with no rescaling.
// ---------------------------------------------------------------------------
__global__ __launch_bounds__(128) void flash_kernel(
    const bf16* __restrict__ Q,      // [B][N][32] bf16
    const bf16* __restrict__ Kp,     // [B][N][32] bf16
    const bf16* __restrict__ V,      // [B][C][N] bf16
    const float* __restrict__ x,     // [B][C][N] f32 (residual)
    const float* __restrict__ gamma,
    float* __restrict__ out)         // [B][C][N] f32
{
  __shared__ __align__(16) unsigned char smem[16384 + 4 * 1024];
  const int tid  = threadIdx.x;
  const int w    = tid >> 5, lane = tid & 31;
  const int lq   = lane & 15, hi = lane >> 4;
  const int b    = blockIdx.x >> 6;              // 64 blocks per batch
  const int qt   = (blockIdx.x & 63) * 4 + w;
  const int n0   = qt * 16;

  unsigned char* vlds = smem;                                // [256 c][64 B]
  __bf16*        plds = (__bf16*)(smem + 16384 + w * 1024);  // [16][32] bf16

  // Q fragment (A-matrix 16x32)
  BF16x16 qa;
  {
    const bf16* qrow = Q + ((size_t)b * NN + (n0 + lq)) * DQ;
    qa.h[0] = *(const v8bf*)(qrow + hi * 8);
    qa.h[1] = *(const v8bf*)(qrow + 16 + hi * 8);
  }
  const bf16* kbase = Kp + (size_t)b * NN * DQ;
  const bf16* vbase = V  + (size_t)b * CCH * NN;

  // ---------------- Pass 1: row maxima ----------------
  float mrow[8];
  #pragma unroll
  for (int r = 0; r < 8; ++r) mrow[r] = -INFINITY;

  for (int j = 0; j < NN / 32; ++j) {
    const bf16* krow = kbase + (size_t)(j * 32 + lq) * DQ;
    BF16x16 kf0, kf1;
    kf0.h[0] = *(const v8bf*)(krow + hi * 8);
    kf0.h[1] = *(const v8bf*)(krow + 16 + hi * 8);
    kf1.h[0] = *(const v8bf*)(krow + 16 * DQ + hi * 8);
    kf1.h[1] = *(const v8bf*)(krow + 16 * DQ + 16 + hi * 8);
    v8f s0 = {}, s1 = {};
    s0 = wmma_bf16(qa.v, kf0.v, s0);
    s1 = wmma_bf16(qa.v, kf1.v, s1);
    #pragma unroll
    for (int r = 0; r < 8; ++r)
      mrow[r] = fmaxf(mrow[r], fmaxf(s0[r], s1[r]));
  }
  // sc folds 1/sqrt(DQK) and log2(e) into a single fma before v_exp
  const float sc = 0.17677669529663688f * 1.4426950408889634f;
  float mb[8];
  #pragma unroll
  for (int r = 0; r < 8; ++r) mb[r] = hmax16(mrow[r]) * sc;

  // ---------------- Pass 2: P, l, O ----------------
  v8f Oacc[16];
  #pragma unroll
  for (int i = 0; i < 16; ++i) Oacc[i] = (v8f){};
  float lsum[8];
  #pragma unroll
  for (int r = 0; r < 8; ++r) lsum[r] = 0.f;

  for (int j = 0; j < NN / 32; ++j) {
    const int key0 = j * 32;

    __syncthreads();                              // WAR on V tile
    #pragma unroll
    for (int t = 0; t < 8; ++t) {                 // 128 thr x 8 = 1024 b128s
      const int idx = tid + t * 128;
      const int row = idx >> 2, seg = idx & 3;
      stage16(vbase + (size_t)row * NN + key0 + seg * 8,
              vlds + row * 64 + seg * 16);
    }
    wait_stage();                                 // own wave's ASYNCcnt -> 0
    __syncthreads();                              // all waves' data visible

    const bf16* krow = kbase + (size_t)(key0 + lq) * DQ;
    BF16x16 kf0, kf1;
    kf0.h[0] = *(const v8bf*)(krow + hi * 8);
    kf0.h[1] = *(const v8bf*)(krow + 16 + hi * 8);
    kf1.h[0] = *(const v8bf*)(krow + 16 * DQ + hi * 8);
    kf1.h[1] = *(const v8bf*)(krow + 16 * DQ + 16 + hi * 8);
    v8f s0 = {}, s1 = {};
    s0 = wmma_bf16(qa.v, kf0.v, s0);
    s1 = wmma_bf16(qa.v, kf1.v, s1);

    #pragma unroll
    for (int r = 0; r < 8; ++r) {
      const float p0 = __builtin_exp2f(__builtin_fmaf(s0[r], sc, -mb[r]));
      const float p1 = __builtin_exp2f(__builtin_fmaf(s1[r], sc, -mb[r]));
      lsum[r] += p0 + p1;
      const int prow = r + 8 * hi;
      plds[prow * 32 + lq]      = (__bf16)p0;     // cols 0..15
      plds[prow * 32 + lq + 16] = (__bf16)p1;     // cols 16..31
    }
    asm volatile("s_wait_dscnt 0" ::: "memory");

    // Reload P in A-fragment layout (2x ds_load_b128)
    BF16x16 pf;
    {
      const __bf16* pr = plds + lq * 32;
      pf.h[0] = *(const v8bf*)(pr + hi * 8);
      pf.h[1] = *(const v8bf*)(pr + 16 + hi * 8);
    }
    // O += P @ V over 16 c-tiles; V B-fragments straight from LDS rows
    #pragma unroll
    for (int cc = 0; cc < 16; ++cc) {
      BF16x16 vf;
      const unsigned char* vr = vlds + (cc * 16 + lq) * 64;
      vf.h[0] = *(const v8bf*)(vr + hi * 16);
      vf.h[1] = *(const v8bf*)(vr + 32 + hi * 16);
      Oacc[cc] = wmma_bf16(pf.v, vf.v, Oacc[cc]);
    }
  }

  // Epilogue: out = gamma * (O / l) + x
  float rinv[8];
  #pragma unroll
  for (int r = 0; r < 8; ++r) rinv[r] = 1.0f / hsum16(lsum[r]);
  const float g = gamma[0];
  const float* xb = x   + (size_t)b * CCH * NN;
  float*       ob = out + (size_t)b * CCH * NN;
  #pragma unroll
  for (int cc = 0; cc < 16; ++cc) {
    const int c = cc * 16 + lq;
    #pragma unroll
    for (int r = 0; r < 8; ++r) {
      const int n = n0 + r + 8 * hi;
      const size_t idx = (size_t)c * NN + n;
      ob[idx] = g * (Oacc[cc][r] * rinv[r]) + xb[idx];
    }
  }
}

// ---------------------------------------------------------------------------
extern "C" void kernel_launch(void* const* d_in, const int* in_sizes, int n_in,
                              void* d_out, int out_size, void* d_ws, size_t ws_size,
                              hipStream_t stream) {
  (void)in_sizes; (void)n_in; (void)out_size; (void)ws_size;
  const float* x     = (const float*)d_in[0];
  const float* Wq    = (const float*)d_in[1];
  const float* bq    = (const float*)d_in[2];
  const float* Wk    = (const float*)d_in[3];
  const float* bk    = (const float*)d_in[4];
  const float* Wv    = (const float*)d_in[5];
  const float* bv    = (const float*)d_in[6];
  const float* gamma = (const float*)d_in[7];
  float* out = (float*)d_out;

  bf16* Qb = (bf16*)d_ws;                         // [B][N][32]  1 MB
  bf16* Kb = Qb + (size_t)BB * NN * DQ;           // [B][N][32]  1 MB
  bf16* Vb = Kb + (size_t)BB * NN * DQ;           // [B][C][N]   8 MB

  // Q/K: B*(N/16)*(32/16) = 2048 wave-tiles; 8 waves per 256-thread block
  proj_kernel<0><<<256,  256, 0, stream>>>(x, Wq, bq, Qb, DQ,  2048);
  proj_kernel<0><<<256,  256, 0, stream>>>(x, Wk, bk, Kb, DQ,  2048);
  // V: B*(N/16)*(256/16) = 16384 wave-tiles
  proj_kernel<1><<<2048, 256, 0, stream>>>(x, Wv, bv, Vb, CCH, 16384);
  // Flash attention: 1024 wave-tiles, 4 waves/block sharing the V stage
  flash_kernel<<<256, 128, 0, stream>>>(Qb, Kb, Vb, x, gamma, out);
}
